// NTXent_14233521619132
// MI455X (gfx1250) — compile-verified, matrix operand order checked by reading, not run on hardware
//
#include <hip/hip_runtime.h>
#include <hip/hip_bf16.h>

typedef __attribute__((ext_vector_type(16))) _Float16 v16h;
typedef __attribute__((ext_vector_type(8)))  _Float16 v8h;
typedef __attribute__((ext_vector_type(2)))  _Float16 v2h;
typedef __attribute__((ext_vector_type(8)))  float    v8f;

#define NROWS 8192   // 2*BATCH
#define DIM   512
#define HALF  4096   // BATCH

#define BM 256       // block tile (rows == cols)
#define BK 32        // K step (matches WMMA K)
#define LDSW 40      // padded LDS row stride in f16 (80B -> conflict-free)
#define SLAB (BM * LDSW)          // one A or B slab, in f16 elements
#define BUFE (2 * SLAB)           // one double-buffer stage (A+B), elements
#define LDS_BYTES (2 * BUFE * 2)  // 81920 bytes total

// ---------------------------------------------------------------------------
// Kernel 1: per-row L2 norm, write f16 zhat = z * invnorm * sqrt(1/TEMP).
// One block (256 threads) per row; also zeroes rowsum[row] for this call.
// ---------------------------------------------------------------------------
__global__ __launch_bounds__(256) void ntxent_normalize(
    const float* __restrict__ z_i, const float* __restrict__ z_j,
    _Float16* __restrict__ zhat, float* __restrict__ rowsum) {
  const int row = blockIdx.x;
  const int t   = threadIdx.x;
  const float* src = (row < HALF) ? (z_i + (size_t)row * DIM)
                                  : (z_j + (size_t)(row - HALF) * DIM);
  float2 v = ((const float2*)src)[t];
  float ss = v.x * v.x + v.y * v.y;

  __shared__ float sdata[256];
  sdata[t] = ss;
  __syncthreads();
  for (int off = 128; off > 0; off >>= 1) {
    if (t < off) sdata[t] += sdata[t + off];
    __syncthreads();
  }
  // scale = invnorm * sqrt(1/TEMP); TEMP = 0.5 -> sqrt(2)
  const float scale = 1.41421356237f * rsqrtf(fmaxf(sdata[0], 1e-30f));
  __syncthreads();

  v2h h;
  h.x = (_Float16)(v.x * scale);
  h.y = (_Float16)(v.y * scale);
  ((v2h*)(zhat + (size_t)row * DIM))[t] = h;

  if (t == 0) rowsum[row] = 0.0f;   // re-init accumulator every call
}

// ---------------------------------------------------------------------------
// Async global -> LDS 16-byte copy (wave-wide, per-lane addresses).
// vdst VGPR carries the LDS byte address, vaddr pair the global address.
// Tracked by ASYNCcnt; completions are in-order -> partial waits are safe.
// ---------------------------------------------------------------------------
__device__ __forceinline__ void async_copy16(unsigned lds_off, const void* g) {
  asm volatile("global_load_async_to_lds_b128 %0, %1, off"
               :: "v"(lds_off), "v"((unsigned long long)(uintptr_t)g)
               : "memory");
}

// ---------------------------------------------------------------------------
// Kernel 2: 256x256 sim block per workgroup (16 waves, 64x64 each).
// Double-buffered LDS slabs filled by async-to-LDS DMA overlapping the
// 4x4 V_WMMA_F32_16X16X32_F16 compute; fused exp + diag mask + row sums.
// ---------------------------------------------------------------------------
__global__ __launch_bounds__(512) void ntxent_gemm_rowsum(
    const _Float16* __restrict__ zhat, float* __restrict__ rowsum) {
  extern __shared__ _Float16 lds[];    // [2][A|B][BM][LDSW] = 80 KB

  const int lane = threadIdx.x;        // 0..31
  const int w    = threadIdx.y;        // wave 0..15
  const int m0   = lane & 15;
  const int hi   = lane >> 4;          // lane half selects K subgroup
  const int wm   = w >> 2;             // wave's 64-row strip in block
  const int wn   = w & 3;              // wave's 64-col strip in block
  const int tid  = w * 32 + lane;      // 0..511
  const int bi   = blockIdx.y;
  const int bj   = blockIdx.x;

  const _Float16* Ag = zhat + (size_t)(BM * bi) * DIM;
  const _Float16* Bg = zhat + (size_t)(BM * bj) * DIM;

  // this thread's two 16B staging slots (1024 slots per slab)
  const unsigned row0   = (unsigned)tid >> 2;          // slot tid
  const unsigned chunk0 = (unsigned)tid & 3;
  const unsigned row1   = (unsigned)(tid + 512) >> 2;  // slot tid+512
  const unsigned chunk1 = (unsigned)(tid + 512) & 3;
  const unsigned ldsbase = (unsigned)(uintptr_t)lds;

  // stage slabs for K-step `k` into double-buffer stage `buf`
  auto stage = [&](int buf, int k) {
    const unsigned abase = ldsbase + (unsigned)buf * (BUFE * 2);
    const unsigned bbase = abase + SLAB * 2;
    async_copy16(abase + (row0 * LDSW + chunk0 * 8) * 2,
                 Ag + (size_t)row0 * DIM + k + chunk0 * 8);
    async_copy16(bbase + (row0 * LDSW + chunk0 * 8) * 2,
                 Bg + (size_t)row0 * DIM + k + chunk0 * 8);
    async_copy16(abase + (row1 * LDSW + chunk1 * 8) * 2,
                 Ag + (size_t)row1 * DIM + k + chunk1 * 8);
    async_copy16(bbase + (row1 * LDSW + chunk1 * 8) * 2,
                 Bg + (size_t)row1 * DIM + k + chunk1 * 8);
  };

  v8f acc[4][4] = {};

  stage(0, 0);                         // prologue: fill buffer 0

  int buf = 0;
  for (int k = 0; k < DIM; k += BK, buf ^= 1) {
    const int kn = k + BK;
    if (kn < DIM) {
      stage(buf ^ 1, kn);              // DMA next slab during compute
      asm volatile("s_wait_asynccnt 0x4" ::: "memory");  // current slab done
    } else {
      asm volatile("s_wait_asynccnt 0x0" ::: "memory");
    }
    __syncthreads();

    const _Float16* Asb = lds + buf * BUFE;
    const _Float16* Bsb = Asb + SLAB;

    // ---- fragment loads from LDS + 16 WMMA ----
    v16h a[4], b[4];
#pragma unroll
    for (int t = 0; t < 4; ++t) {
      // 16-bit A-matrix 16x32 layout: lanes 0-15 hold K=[0..7]+[16..23],
      // lanes 16-31 hold K=[8..15]+[24..31], row M = lane%16.
      const _Float16* ar = Asb + (64 * wm + 16 * t + m0) * LDSW + hi * 8;
      v8h alo = *(const v8h*)ar;
      v8h ahi = *(const v8h*)(ar + 16);
      a[t] = __builtin_shufflevector(alo, ahi,
             0, 1, 2, 3, 4, 5, 6, 7, 8, 9, 10, 11, 12, 13, 14, 15);
      const _Float16* br = Bsb + (64 * wn + 16 * t + m0) * LDSW + hi * 8;
      v8h blo = *(const v8h*)br;
      v8h bhi = *(const v8h*)(br + 16);
      b[t] = __builtin_shufflevector(blo, bhi,
             0, 1, 2, 3, 4, 5, 6, 7, 8, 9, 10, 11, 12, 13, 14, 15);
    }
#pragma unroll
    for (int mt = 0; mt < 4; ++mt)
#pragma unroll
      for (int nt = 0; nt < 4; ++nt)
        acc[mt][nt] = __builtin_amdgcn_wmma_f32_16x16x32_f16(
            /*neg_a=*/false, a[mt], /*neg_b=*/false, b[nt],
            /*c_mod=*/(short)0, acc[mt][nt],
            /*reuse_a=*/false, /*reuse_b=*/false);

    __syncthreads();   // all reads done before this buffer is refilled
  }

  // ---- epilogue: C tile layout: VGPR v, lanes 0-15 -> M=v, N=lane;
  //      lanes 16-31 -> M=v+8, N=lane-16.
  const int gi0 = BM * bi + 64 * wm;
  const int gj0 = BM * bj + 64 * wn + m0;
#pragma unroll
  for (int mt = 0; mt < 4; ++mt) {
#pragma unroll
    for (int v = 0; v < 8; ++v) {
      const int gi = gi0 + 16 * mt + v + 8 * hi;
      float s = 0.0f;
#pragma unroll
      for (int nt = 0; nt < 4; ++nt) {
        const int gj = gj0 + 16 * nt;
        float e = __expf(acc[mt][nt][v]);     // sim <= 2 -> exp <= e^2, safe
        s += (gi == gj) ? 0.0f : e;           // mask self-similarity diagonal
      }
#pragma unroll
      for (int off = 1; off < 16; off <<= 1)
        s += __shfl_xor(s, off, 32);          // reduce 16 cols per half-wave
      if (m0 == 0) atomicAdd(&rowsum[gi], s);
    }
  }
}

// ---------------------------------------------------------------------------
// Kernel 3: positive-pair logits pos[i] = dot(zhat[i], zhat[(i+N) % 2N]).
// ---------------------------------------------------------------------------
__global__ __launch_bounds__(32) void ntxent_pos(
    const _Float16* __restrict__ zhat, float* __restrict__ pos) {
  const int i = blockIdx.x;
  const int j = (i + HALF) & (NROWS - 1);
  const int lane = threadIdx.x;
  const _Float16* pi = zhat + (size_t)i * DIM + lane * 16;
  const _Float16* pj = zhat + (size_t)j * DIM + lane * 16;
  v8h a0 = *(const v8h*)pi;
  v8h a1 = *(const v8h*)(pi + 8);
  v8h b0 = *(const v8h*)pj;
  v8h b1 = *(const v8h*)(pj + 8);
  float s = 0.0f;
#pragma unroll
  for (int e = 0; e < 8; ++e) {
    s += (float)a0[e] * (float)b0[e];
    s += (float)a1[e] * (float)b1[e];
  }
#pragma unroll
  for (int off = 1; off < 32; off <<= 1)
    s += __shfl_xor(s, off, 32);
  if (lane == 0) pos[i] = s;
}

// ---------------------------------------------------------------------------
// Kernel 4: loss = mean_i( log(rowsum[i]) - pos[i] ).  One block.
// ---------------------------------------------------------------------------
__global__ __launch_bounds__(256) void ntxent_finalize(
    const float* __restrict__ rowsum, const float* __restrict__ pos,
    float* __restrict__ out) {
  const int t = threadIdx.x;
  float s = 0.0f;
  for (int i = t; i < NROWS; i += 256)
    s += __logf(rowsum[i]) - pos[i];
  __shared__ float sdata[256];
  sdata[t] = s;
  __syncthreads();
  for (int off = 128; off > 0; off >>= 1) {
    if (t < off) sdata[t] += sdata[t + off];
    __syncthreads();
  }
  if (t == 0) out[0] = sdata[0] * (1.0f / NROWS);
}

// ---------------------------------------------------------------------------
extern "C" void kernel_launch(void* const* d_in, const int* in_sizes, int n_in,
                              void* d_out, int out_size, void* d_ws, size_t ws_size,
                              hipStream_t stream) {
  const float* z_i = (const float*)d_in[0];
  const float* z_j = (const float*)d_in[1];
  float* out = (float*)d_out;

  char* ws = (char*)d_ws;
  _Float16* zhat   = (_Float16*)ws;                                  // 8 MiB
  float*    rowsum = (float*)(ws + (size_t)NROWS * DIM * sizeof(_Float16));
  float*    pos    = rowsum + NROWS;

  ntxent_normalize<<<NROWS, 256, 0, stream>>>(z_i, z_j, zhat, rowsum);
  ntxent_gemm_rowsum<<<dim3(NROWS / BM, NROWS / BM), dim3(32, 16),
                       LDS_BYTES, stream>>>(zhat, rowsum);
  ntxent_pos<<<NROWS, 32, 0, stream>>>(zhat, pos);
  ntxent_finalize<<<1, 256, 0, stream>>>(rowsum, pos, out);
}
